// ST_MAN_24541443129672
// MI455X (gfx1250) — matrix operand order repaired and checked by reading, not compiled, not run on hardware
//
#include <hip/hip_runtime.h>
#include <hip/hip_bf16.h>
#include <math.h>
#include <string.h>

// ---------------------------------------------------------------------------
// ST-MAN forward for MI455X (gfx1250, wave32, WMMA).
// f16 operands, f32 accumulation via v_wmma_f32_16x16x32_f16.
// Block tile 32(M) x 128(N), K-chunk 32, LDS double-buffered; each wave owns a
// 16x32 C tile (2 accumulators, shared A fragment). STE broadcast tensor is
// factored (se[n] + te[b,s]) and fused into the qkv GEMM epilogue.
// ---------------------------------------------------------------------------

typedef _Float16 h16;
typedef __attribute__((ext_vector_type(16))) _Float16 v16h;
typedef __attribute__((ext_vector_type(8)))  float    v8f;

#define DEVI __device__ __forceinline__

DEVI float geluf(float x) { return 0.5f * x * (1.0f + erff(x * 0.7071067811865475f)); }

// Model constants
#define PP 12
#define NNODE 325
#define DD 512
#define KH 8
#define BB 8
#define SENC 36
#define TOK_ENC (BB * SENC * NNODE)   // 93600
#define TOK_DEC (BB * PP * NNODE)     // 31200

struct GP {
  const h16* A; const h16* B; const float* bias;
  h16* C; const h16* RES; const h16* steS; const h16* steT;
  int M, N, Kdim, Ka, Kb, Nb;
  int lda, ldb, ldc, ldr;
  int Z2, Z3;
  long long sA1,sA2,sA3, sB1,sB2,sB3, sC1,sC2,sC3, sR1,sR2,sR3;
  int NN, S, te_off, Nreal; float scale;
};

// EPI: 0 plain(+bias) 1 gelu(+bias) 2 qkv(+bias+STE,gelu) 3 emb(+bias+posenc)
//      4 scores(scale,mask) 5 pv(+res) 6 ffn2(+bias+res)
template<int EPI>
DEVI void epi_store(const GP& p, const h16* Rb, h16* Cb, v8f acc, int rbase, int colg) {
  if (colg >= p.N) return;
#pragma unroll
  for (int i = 0; i < 8; ++i) {
    int rowg = rbase + i;
    if (rowg >= p.M) continue;
    float v = acc[i];
    if (EPI != 4 && EPI != 5 && p.bias) v += p.bias[colg];
    if (EPI == 1) {
      v = geluf(v);
    } else if (EPI == 2) {
      int n  = rowg % p.NN;
      int s  = (rowg / p.NN) % p.S;
      int b  = rowg / (p.NN * p.S);
      int tr = b * 24 + p.te_off + (s % 12);
      v += (float)p.steS[(size_t)n  * p.N + colg];
      v += (float)p.steT[(size_t)tr * p.N + colg];
      v = geluf(v);
    } else if (EPI == 3) {
      int pos = rowg / p.NN;
      int ii  = colg & ~1;
      float ang = (float)pos * __powf(10000.0f, -(float)ii / 512.0f);
      v += (colg & 1) ? cosf(ang) : sinf(ang);
    } else if (EPI == 4) {
      v = (colg < p.Nreal) ? v * p.scale : -30000.0f;
    } else if (EPI == 5 || EPI == 6) {
      v += (float)Rb[(size_t)rowg * p.ldr + colg];
    }
    Cb[(size_t)rowg * p.ldc + colg] = (h16)v;
  }
}

template<int EPI, bool BT>
__global__ __launch_bounds__(256) void gemm_k(GP p) {
  // A panels: 32 rows x 32 halves, padded to 40 halves/row (conflict-free frags)
  __shared__ unsigned int sA[2][32 * 20];
  // B panels swizzled as K-pairs: 16 rows x (128 col-dwords + 4 pad)
  __shared__ unsigned int sB[2][16 * 132];

  const int tid = threadIdx.x;
  const int z  = blockIdx.z;
  const int z3 = z % p.Z3;
  const int zt = z / p.Z3;
  const int z2 = zt % p.Z2;
  const int z1 = zt / p.Z2;

  const h16* Ab = p.A + (size_t)z1*p.sA1 + (size_t)z2*p.sA2 + (size_t)z3*p.sA3;
  const h16* Bb = p.B + (size_t)z1*p.sB1 + (size_t)z2*p.sB2 + (size_t)z3*p.sB3;
  h16*       Cb = p.C + (size_t)z1*p.sC1 + (size_t)z2*p.sC2 + (size_t)z3*p.sC3;
  const h16* Rb = (EPI == 5 || EPI == 6)
      ? p.RES + (size_t)z1*p.sR1 + (size_t)z2*p.sR2 + (size_t)z3*p.sR3 : (const h16*)0;

  const int m0 = blockIdx.x * 32;
  const int n0 = blockIdx.y * 128;
  const int w = tid >> 5, l = tid & 31;
  const int wm = w >> 2, wn = w & 3;

  unsigned int ra[2], rb[8];

  // Fetch one K-chunk into registers (clamped addresses + select, no branches).
  auto fetch = [&](int k0, bool have) {
#pragma unroll
    for (int j = 0; j < 2; ++j) {
      int i = tid + 256 * j;
      int m = i >> 4, kk = (i & 15) << 1;
      int mg = m0 + m, kg = k0 + kk;
      bool ok = have & (mg < p.M) & (kg < p.Ka);
      int mgc = mg < p.M ? mg : (p.M - 1);
      int kgc = kg < p.Ka ? kg : 0;
      unsigned int v = *reinterpret_cast<const unsigned int*>(Ab + (size_t)mgc * p.lda + kgc);
      ra[j] = ok ? v : 0u;
    }
    if (!BT) {
      // B[k][n], dword along n (Nb is even for all normal-mode GEMMs here)
#pragma unroll
      for (int j = 0; j < 8; ++j) {
        int i = tid + 256 * j;
        int k = i >> 6, n = (i & 63) << 1;
        int kg = k0 + k, ng = n0 + n;
        bool ok = have & (kg < p.Kb) & (ng < p.Nb);
        int kgc = kg < p.Kb ? kg : 0;
        int ngc = ng < p.Nb ? ng : 0;
        unsigned int v = *reinterpret_cast<const unsigned int*>(Bb + (size_t)kgc * p.ldb + ngc);
        rb[j] = ok ? v : 0u;
      }
    } else {
      // B[k][n] = src[n*ldb + k] (transposed source), dword along k
#pragma unroll
      for (int j = 0; j < 8; ++j) {
        int i = tid + 256 * j;
        int n = i >> 4, kk = (i & 15) << 1;
        int ng = n0 + n, kg = k0 + kk;
        bool ok = have & (ng < p.Nb) & (kg < p.Kb);
        int ngc = ng < p.Nb ? ng : 0;
        int kgc = kg < p.Kb ? kg : 0;
        unsigned int v = *reinterpret_cast<const unsigned int*>(Bb + (size_t)ngc * p.ldb + kgc);
        rb[j] = ok ? v : 0u;
      }
    }
  };

  auto stage = [&](int buf) {
#pragma unroll
    for (int j = 0; j < 2; ++j) {
      int i = tid + 256 * j;
      int m = i >> 4, kk = (i & 15) << 1;
      sA[buf][m * 20 + (kk >> 1)] = ra[j];
    }
    if (!BT) {
      h16* sh = reinterpret_cast<h16*>(sB[buf]);
#pragma unroll
      for (int j = 0; j < 8; ++j) {
        int i = tid + 256 * j;
        int k = i >> 6, n = (i & 63) << 1;
        int k2 = k >> 1, r = k & 1;
        union { unsigned int u; h16 h[2]; } t; t.u = rb[j];
        sh[k2 * 264 + (n << 1) + r]       = t.h[0];
        sh[k2 * 264 + ((n + 1) << 1) + r] = t.h[1];
      }
    } else {
#pragma unroll
      for (int j = 0; j < 8; ++j) {
        int i = tid + 256 * j;
        int n = i >> 4, kk = (i & 15) << 1;
        sB[buf][(kk >> 1) * 132 + n] = rb[j];
      }
    }
  };

  v8f acc0 = {}, acc1 = {};
  const int kch = (p.Kdim + 31) >> 5;

  fetch(0, true);
  stage(0);
  __syncthreads();

  int cur = 0;
  for (int kc = 0; kc < kch; ++kc) {
    const bool have = (kc + 1) < kch;
    fetch((kc + 1) << 5, have);
    if ((kc + 2) < kch)   // speculative L2 prefetch of the chunk after next
      __builtin_prefetch((const void*)(Ab + (size_t)m0 * p.lda + ((kc + 2) << 5)), 0, 0);

    union { v16h h; unsigned int u[8]; } af, bf0, bf1;
    {
      int r = wm * 16 + (l & 15), sel = l >> 4;
#pragma unroll
      for (int j = 0; j < 4; ++j) {
        af.u[j]     = sA[cur][r * 20 + sel * 8 + j];       // K = sel*16 + 2j
        af.u[4 + j] = sA[cur][r * 20 + sel * 8 + 4 + j];   // K = sel*16 + 8 + 2j
      }
    }
    {
      int nb = wn * 32 + (l & 15), kh = l >> 4;
#pragma unroll
      for (int j = 0; j < 8; ++j) {
        bf0.u[j] = sB[cur][(kh * 8 + j) * 132 + nb];        // K = kh*16 + 2j
        bf1.u[j] = sB[cur][(kh * 8 + j) * 132 + nb + 16];
      }
    }
    acc0 = __builtin_amdgcn_wmma_f32_16x16x32_f16(false, af.h, false, bf0.h,
                                                  (short)0, acc0, false, false);
    acc1 = __builtin_amdgcn_wmma_f32_16x16x32_f16(false, af.h, false, bf1.h,
                                                  (short)0, acc1, false, false);
    if (have) stage(cur ^ 1);
    __syncthreads();
    cur ^= 1;
  }

  const int colg  = n0 + wn * 32 + (l & 15);
  const int rbase = m0 + wm * 16 + ((l >> 4) << 3);
  epi_store<EPI>(p, Rb, Cb, acc0, rbase, colg);
  epi_store<EPI>(p, Rb, Cb, acc1, rbase, colg + 16);
}

// In-place row softmax over f16 logits (padded cols hold -30000 -> prob 0).
__global__ __launch_bounds__(128) void softmax_k(h16* buf, int rows, int Np) {
  int wid = threadIdx.x >> 5, l = threadIdx.x & 31;
  int r = blockIdx.x * 4 + wid;
  if (r >= rows) return;
  h16* row = buf + (size_t)r * Np;
  float vals[11];
  int cnt = 0;
  float mx = -3.0e38f;
  for (int c = l; c < Np; c += 32) { float v = (float)row[c]; vals[cnt++] = v; if (v > mx) mx = v; }
  for (int o = 16; o; o >>= 1) { float t = __shfl_xor(mx, o, 32); if (t > mx) mx = t; }
  float s = 0.f;
  for (int i = 0; i < cnt; ++i) { float e = __expf(vals[i] - mx); vals[i] = e; s += e; }
  for (int o = 16; o; o >>= 1) s += __shfl_xor(s, o, 32);
  float inv = 1.0f / s;
  cnt = 0;
  for (int c = l; c < Np; c += 32) row[c] = (h16)(vals[cnt++] * inv);
}

__global__ void cvt_k(const float* s, h16* d, int n) {
  int i = blockIdx.x * blockDim.x + threadIdx.x;
  if (i < n) d[i] = (h16)s[i];
}

// First embedding layer: out[t,j] = gelu(x[t,c] * W[j] + b[j])  (rank-1 GEMM)
__global__ void emb_first_k(const float* x, const float* W, const float* b, h16* out, int c) {
  int i = blockIdx.x * blockDim.x + threadIdx.x;
  if (i >= TOK_DEC * DD) return;
  int t = i >> 9, j = i & 511;
  out[i] = (h16)geluf(x[t * 3 + c] * W[j] + b[j]);
}

// te first layer: one-hot concat @ W == sum of 5 selected rows of W
__global__ void te_first_k(const int* TE, const float* W, const float* b, h16* out) {
  int i = blockIdx.x * blockDim.x + threadIdx.x;
  if (i >= BB * 24 * DD) return;
  int bs = i >> 9, j = i & 511;
  const int* e = TE + bs * 5;
  float v = W[(size_t)(e[0])       * DD + j] + W[(size_t)(7   + e[1]) * DD + j]
          + W[(size_t)(295 + e[2]) * DD + j] + W[(size_t)(299 + e[3]) * DD + j]
          + W[(size_t)(302 + e[4]) * DD + j] + b[j];
  out[i] = (h16)geluf(v);
}

// h = max_c (xcat[c] + enc_out[c])
__global__ void hmax_k(const h16* xcat, const h16* enc, h16* h) {
  int i = blockIdx.x * blockDim.x + threadIdx.x;
  if (i >= TOK_DEC * DD) return;
  int t = i >> 9, j = i & 511;
  int b = t / (PP * NNODE), rem = t % (PP * NNODE);
  int pp = rem / NNODE, n = rem % NNODE;
  float mx = -3.0e38f;
#pragma unroll
  for (int c = 0; c < 3; ++c) {
    size_t a = ((size_t)(b * SENC + c * PP + pp) * NNODE + n) * DD + j;
    float v = (float)xcat[a] + (float)enc[a];
    if (v > mx) mx = v;
  }
  h[i] = (h16)mx;
}

__global__ void add_k(const h16* a, const h16* b, h16* o, int n) {
  int i = blockIdx.x * blockDim.x + threadIdx.x;
  if (i < n) o[i] = (h16)((float)a[i] + (float)b[i]);
}

// Final D->1 projection: one wave per token.
__global__ __launch_bounds__(256) void out2_k(const h16* o1, const float* W,
                                              const float* b, float* out, int M) {
  int t = blockIdx.x * 8 + (threadIdx.x >> 5);
  int l = threadIdx.x & 31;
  if (t >= M) return;
  const h16* row = o1 + (size_t)t * DD;
  float s = 0.f;
  for (int j = l; j < DD; j += 32) s += (float)row[j] * W[j];
  for (int o = 16; o; o >>= 1) s += __shfl_xor(s, o, 32);
  if (l == 0) out[t] = s + b[0];
}

// ---------------------------------------------------------------------------
static inline GP gp0() { GP g; memset(&g, 0, sizeof(g)); g.Z2 = 1; g.Z3 = 1; return g; }
static inline dim3 ggrid(int M, int N, int Z) { return dim3((M + 31) / 32, (N + 127) / 128, Z); }

extern "C" void kernel_launch(void* const* d_in, const int* in_sizes, int n_in,
                              void* d_out, int out_size, void* d_ws, size_t ws_size,
                              hipStream_t stream) {
  (void)in_sizes; (void)n_in; (void)out_size; (void)ws_size;
  char* ws = (char*)d_ws;
  h16* WH = (h16*)ws;   // static f16 pool at offset 0

  // static pool offsets (in halves)
  const size_t OH_EMBW2[3] = {0, 262144, 524288};
  const size_t OH_FCSW1 = 786432,  OH_FCSW2 = 1048576, OH_FCTW2 = 1310720;
  const size_t OH_QKVE  = 1572864, OH_FFNE1 = 2621440, OH_FFNE2 = 2883584;
  const size_t OH_TQKV  = 3145728, OH_TFFN1 = 4194304, OH_TFFN2 = 4456448;
  const size_t OH_SQKV  = 4718592, OH_SFFN1 = 5767168, OH_SFFN2 = 6029312;
  const size_t OH_OUT1  = 6291456, OH_SE    = 6553600, OH_SE1   = 6720000;
  const size_t OH_SEO   = 6886400, OH_TE1   = 7052800, OH_TEO   = 7151104;
  const size_t OH_SQS[3] = {7249408, 7582208, 7915008};   // se @ Wqkv_bot (325x1024) x3
  const size_t OH_SQT[3] = {8247808, 8444416, 8641024};   // te @ Wqkv_bot (192x1024) x3

  const size_t MB = 1024 * 1024;
  h16* t1      = (h16*)(ws + 32  * MB);   // emb hidden (dead before enc_out)
  h16* enc_out = (h16*)(ws + 32  * MB);
  h16* xcat    = (h16*)(ws + 128 * MB);   // (B,36,NN,512) f16
  h16* qkvE    = (h16*)(ws + 224 * MB);   // (B,36,NN,1024)
  h16* scoresT = (h16*)(ws + 416 * MB);   // 20800 x 36 x 48
  h16* attC    = (h16*)(ws + 512 * MB);
  h16* hid     = (h16*)(ws + 608 * MB);
  // decoder aliases (lifetimes disjoint with the slabs above)
  h16* hbuf    = (h16*)(ws + 416 * MB);
  h16* htb     = (h16*)(ws + 416 * MB + 33 * MB);
  h16* hfin    = (h16*)(ws + 416 * MB + 66 * MB);
  h16* qkv2    = (h16*)(ws + 128 * MB);
  h16* qkv3    = (h16*)(ws + 128 * MB);
  h16* scores2 = (h16*)(ws + 224 * MB);
  h16* attC2   = (h16*)(ws + 224 * MB + 16 * MB);
  h16* scores3 = (h16*)(ws + 224 * MB);
  h16* attC3   = (h16*)(ws + 512 * MB);
  h16* hs      = (h16*)(ws + 512 * MB + 33 * MB);
  h16* o1      = (h16*)(ws + 512 * MB + 66 * MB);

  auto cvt = [&](int idx, size_t off, int n) {
    cvt_k<<<(n + 255) / 256, 256, 0, stream>>>((const float*)d_in[idx], WH + off, n);
  };
  // weights -> f16
  cvt(5,  OH_EMBW2[0], 262144); cvt(9,  OH_EMBW2[1], 262144); cvt(13, OH_EMBW2[2], 262144);
  cvt(15, OH_FCSW1, 262144);    cvt(17, OH_FCSW2, 262144);    cvt(21, OH_FCTW2, 262144);
  cvt(23, OH_QKVE, 1048576);    cvt(25, OH_FFNE1, 262144);    cvt(27, OH_FFNE2, 262144);
  cvt(29, OH_TQKV, 1048576);    cvt(31, OH_TFFN1, 262144);    cvt(33, OH_TFFN2, 262144);
  cvt(35, OH_SQKV, 1048576);    cvt(37, OH_SFFN1, 262144);    cvt(39, OH_SFFN2, 262144);
  cvt(41, OH_OUT1, 262144);     cvt(2,  OH_SE, 166400);

  // ---- se = fc_s(SE) : 325 tokens ----
  { GP g = gp0(); g.A = WH + OH_SE; g.lda = 512; g.B = WH + OH_FCSW1; g.ldb = 512;
    g.bias = (const float*)d_in[16]; g.C = WH + OH_SE1; g.ldc = 512;
    g.M = 325; g.N = 512; g.Nb = 512; g.Kdim = 512; g.Ka = 512; g.Kb = 512;
    gemm_k<1, false><<<ggrid(325, 512, 1), 256, 0, stream>>>(g); }
  { GP g = gp0(); g.A = WH + OH_SE1; g.lda = 512; g.B = WH + OH_FCSW2; g.ldb = 512;
    g.bias = (const float*)d_in[18]; g.C = WH + OH_SEO; g.ldc = 512;
    g.M = 325; g.N = 512; g.Nb = 512; g.Kdim = 512; g.Ka = 512; g.Kb = 512;
    gemm_k<0, false><<<ggrid(325, 512, 1), 256, 0, stream>>>(g); }

  // ---- te = fc_t(one_hot(TE)) : 192 tokens ----
  te_first_k<<<(BB * 24 * DD + 255) / 256, 256, 0, stream>>>(
      (const int*)d_in[1], (const float*)d_in[19], (const float*)d_in[20], WH + OH_TE1);
  { GP g = gp0(); g.A = WH + OH_TE1; g.lda = 512; g.B = WH + OH_FCTW2; g.ldb = 512;
    g.bias = (const float*)d_in[22]; g.C = WH + OH_TEO; g.ldc = 512;
    g.M = 192; g.N = 512; g.Nb = 512; g.Kdim = 512; g.Ka = 512; g.Kb = 512;
    gemm_k<0, false><<<ggrid(192, 512, 1), 256, 0, stream>>>(g); }

  // ---- STE tables: se@Wqkv_bot, te@Wqkv_bot for the 3 qkv matrices ----
  const size_t qkvw[3] = {OH_QKVE, OH_TQKV, OH_SQKV};
  for (int q = 0; q < 3; ++q) {
    { GP g = gp0(); g.A = WH + OH_SEO; g.lda = 512;
      g.B = WH + qkvw[q] + 512 * 1024; g.ldb = 1024;
      g.C = WH + OH_SQS[q]; g.ldc = 1024;
      g.M = 325; g.N = 1024; g.Nb = 1024; g.Kdim = 512; g.Ka = 512; g.Kb = 512;
      gemm_k<0, false><<<ggrid(325, 1024, 1), 256, 0, stream>>>(g); }
    { GP g = gp0(); g.A = WH + OH_TEO; g.lda = 512;
      g.B = WH + qkvw[q] + 512 * 1024; g.ldb = 1024;
      g.C = WH + OH_SQT[q]; g.ldc = 1024;
      g.M = 192; g.N = 1024; g.Nb = 1024; g.Kdim = 512; g.Ka = 512; g.Kb = 512;
      gemm_k<0, false><<<ggrid(192, 1024, 1), 256, 0, stream>>>(g); }
  }

  // ---- embeddings: xcat[b, c*12+p, n, :] = emb_c(x[...,c]) + posenc ----
  for (int c = 0; c < 3; ++c) {
    emb_first_k<<<(TOK_DEC * DD + 255) / 256, 256, 0, stream>>>(
        (const float*)d_in[0], (const float*)d_in[3 + 4 * c],
        (const float*)d_in[4 + 4 * c], t1, c);
    GP g = gp0(); g.A = t1; g.lda = 512; g.sA1 = 3900LL * 512;
    g.B = WH + OH_EMBW2[c]; g.ldb = 512; g.bias = (const float*)d_in[6 + 4 * c];
    g.C = xcat + (size_t)c * 12 * NNODE * 512; g.ldc = 512; g.sC1 = 36LL * NNODE * 512;
    g.M = 3900; g.N = 512; g.Nb = 512; g.Kdim = 512; g.Ka = 512; g.Kb = 512; g.NN = NNODE;
    gemm_k<3, false><<<ggrid(3900, 512, 8), 256, 0, stream>>>(g);
  }

  // ================= ENCODER =================
  { // qkv = gelu(xcat@Wtop + bias + steS[n] + steT[b,s%12])
    GP g = gp0(); g.A = xcat; g.lda = 512; g.B = WH + OH_QKVE; g.ldb = 1024;
    g.bias = (const float*)d_in[24]; g.C = qkvE; g.ldc = 1024;
    g.steS = WH + OH_SQS[0]; g.steT = WH + OH_SQT[0];
    g.M = TOK_ENC; g.N = 1024; g.Nb = 1024; g.Kdim = 512; g.Ka = 512; g.Kb = 512;
    g.NN = NNODE; g.S = SENC; g.te_off = 0;
    gemm_k<2, false><<<ggrid(TOK_ENC, 1024, 1), 256, 0, stream>>>(g); }
  { // temporal scores: per head (b,n,k), 36x36 over d=64
    GP g = gp0(); g.A = qkvE; g.lda = NNODE * 1024;
    g.sA1 = 36LL * NNODE * 1024; g.sA2 = 1024; g.sA3 = 128;
    g.B = qkvE; g.ldb = NNODE * 1024; g.sB1 = g.sA1; g.sB2 = 1024; g.sB3 = 128;
    g.C = scoresT; g.ldc = 48; g.sC1 = (long long)NNODE * KH * 36 * 48;
    g.sC2 = (long long)KH * 36 * 48; g.sC3 = 36 * 48;
    g.M = 36; g.N = 48; g.Nreal = 36; g.Nb = 36; g.Kdim = 64; g.Ka = 64; g.Kb = 64;
    g.scale = 0.125f; g.Z2 = NNODE; g.Z3 = KH;
    gemm_k<4, true><<<ggrid(36, 48, BB * NNODE * KH), 256, 0, stream>>>(g); }
  softmax_k<<<(BB * NNODE * KH * 36 + 3) / 4, 128, 0, stream>>>(scoresT, BB * NNODE * KH * 36, 48);
  { // PV + residual x
    GP g = gp0(); g.A = scoresT; g.lda = 48;
    g.sA1 = (long long)NNODE * KH * 36 * 48; g.sA2 = (long long)KH * 36 * 48; g.sA3 = 36 * 48;
    g.B = qkvE + 64; g.ldb = NNODE * 1024;
    g.sB1 = 36LL * NNODE * 1024; g.sB2 = 1024; g.sB3 = 128;
    g.C = attC; g.ldc = NNODE * 512; g.sC1 = 36LL * NNODE * 512; g.sC2 = 512; g.sC3 = 64;
    g.RES = xcat; g.ldr = NNODE * 512; g.sR1 = g.sC1; g.sR2 = 512; g.sR3 = 64;
    g.M = 36; g.N = 64; g.Nb = 64; g.Kdim = 48; g.Ka = 48; g.Kb = 36;
    g.Z2 = NNODE; g.Z3 = KH;
    gemm_k<5, false><<<ggrid(36, 64, BB * NNODE * KH), 256, 0, stream>>>(g); }
  { GP g = gp0(); g.A = attC; g.lda = 512; g.B = WH + OH_FFNE1; g.ldb = 512;
    g.bias = (const float*)d_in[26]; g.C = hid; g.ldc = 512;
    g.M = TOK_ENC; g.N = 512; g.Nb = 512; g.Kdim = 512; g.Ka = 512; g.Kb = 512;
    gemm_k<1, false><<<ggrid(TOK_ENC, 512, 1), 256, 0, stream>>>(g); }
  { GP g = gp0(); g.A = hid; g.lda = 512; g.B = WH + OH_FFNE2; g.ldb = 512;
    g.bias = (const float*)d_in[28]; g.C = enc_out; g.ldc = 512;
    g.RES = attC; g.ldr = 512;
    g.M = TOK_ENC; g.N = 512; g.Nb = 512; g.Kdim = 512; g.Ka = 512; g.Kb = 512;
    gemm_k<6, false><<<ggrid(TOK_ENC, 512, 1), 256, 0, stream>>>(g); }
  hmax_k<<<(TOK_DEC * DD + 255) / 256, 256, 0, stream>>>(xcat, enc_out, hbuf);

  // ================= DECODER: temporal =================
  { GP g = gp0(); g.A = hbuf; g.lda = 512; g.B = WH + OH_TQKV; g.ldb = 1024;
    g.bias = (const float*)d_in[30]; g.C = qkv2; g.ldc = 1024;
    g.steS = WH + OH_SQS[1]; g.steT = WH + OH_SQT[1];
    g.M = TOK_DEC; g.N = 1024; g.Nb = 1024; g.Kdim = 512; g.Ka = 512; g.Kb = 512;
    g.NN = NNODE; g.S = 12; g.te_off = 12;
    gemm_k<2, false><<<ggrid(TOK_DEC, 1024, 1), 256, 0, stream>>>(g); }
  { GP g = gp0(); g.A = qkv2; g.lda = NNODE * 1024;
    g.sA1 = 12LL * NNODE * 1024; g.sA2 = 1024; g.sA3 = 128;
    g.B = qkv2; g.ldb = NNODE * 1024; g.sB1 = g.sA1; g.sB2 = 1024; g.sB3 = 128;
    g.C = scores2; g.ldc = 16; g.sC1 = (long long)NNODE * KH * 12 * 16;
    g.sC2 = (long long)KH * 12 * 16; g.sC3 = 12 * 16;
    g.M = 12; g.N = 16; g.Nreal = 12; g.Nb = 12; g.Kdim = 64; g.Ka = 64; g.Kb = 64;
    g.scale = 0.125f; g.Z2 = NNODE; g.Z3 = KH;
    gemm_k<4, true><<<ggrid(12, 16, BB * NNODE * KH), 256, 0, stream>>>(g); }
  softmax_k<<<(BB * NNODE * KH * 12 + 3) / 4, 128, 0, stream>>>(scores2, BB * NNODE * KH * 12, 16);
  { GP g = gp0(); g.A = scores2; g.lda = 16;
    g.sA1 = (long long)NNODE * KH * 12 * 16; g.sA2 = (long long)KH * 12 * 16; g.sA3 = 12 * 16;
    g.B = qkv2 + 64; g.ldb = NNODE * 1024;
    g.sB1 = 12LL * NNODE * 1024; g.sB2 = 1024; g.sB3 = 128;
    g.C = attC2; g.ldc = NNODE * 512; g.sC1 = 12LL * NNODE * 512; g.sC2 = 512; g.sC3 = 64;
    g.RES = hbuf; g.ldr = NNODE * 512; g.sR1 = g.sC1; g.sR2 = 512; g.sR3 = 64;
    g.M = 12; g.N = 64; g.Nb = 64; g.Kdim = 16; g.Ka = 16; g.Kb = 12;
    g.Z2 = NNODE; g.Z3 = KH;
    gemm_k<5, false><<<ggrid(12, 64, BB * NNODE * KH), 256, 0, stream>>>(g); }
  { GP g = gp0(); g.A = attC2; g.lda = 512; g.B = WH + OH_TFFN1; g.ldb = 512;
    g.bias = (const float*)d_in[32]; g.C = hid; g.ldc = 512;
    g.M = TOK_DEC; g.N = 512; g.Nb = 512; g.Kdim = 512; g.Ka = 512; g.Kb = 512;
    gemm_k<1, false><<<ggrid(TOK_DEC, 512, 1), 256, 0, stream>>>(g); }
  { GP g = gp0(); g.A = hid; g.lda = 512; g.B = WH + OH_TFFN2; g.ldb = 512;
    g.bias = (const float*)d_in[34]; g.C = htb; g.ldc = 512;
    g.RES = attC2; g.ldr = 512;
    g.M = TOK_DEC; g.N = 512; g.Nb = 512; g.Kdim = 512; g.Ka = 512; g.Kb = 512;
    gemm_k<6, false><<<ggrid(TOK_DEC, 512, 1), 256, 0, stream>>>(g); }

  // ================= DECODER: spatial =================
  { GP g = gp0(); g.A = htb; g.lda = 512; g.B = WH + OH_SQKV; g.ldb = 1024;
    g.bias = (const float*)d_in[36]; g.C = qkv3; g.ldc = 1024;
    g.steS = WH + OH_SQS[2]; g.steT = WH + OH_SQT[2];
    g.M = TOK_DEC; g.N = 1024; g.Nb = 1024; g.Kdim = 512; g.Ka = 512; g.Kb = 512;
    g.NN = NNODE; g.S = 12; g.te_off = 12;
    gemm_k<2, false><<<ggrid(TOK_DEC, 1024, 1), 256, 0, stream>>>(g); }
  { // spatial scores: per (b,p,k) a 325x325 over d=64
    GP g = gp0(); g.A = qkv3; g.lda = 1024;
    g.sA1 = 12LL * NNODE * 1024; g.sA2 = (long long)NNODE * 1024; g.sA3 = 128;
    g.B = qkv3; g.ldb = 1024; g.sB1 = g.sA1; g.sB2 = g.sA2; g.sB3 = 128;
    g.C = scores3; g.ldc = 336;
    g.sC1 = 12LL * KH * 325 * 336; g.sC2 = (long long)KH * 325 * 336; g.sC3 = 325 * 336;
    g.M = 325; g.N = 336; g.Nreal = 325; g.Nb = 325; g.Kdim = 64; g.Ka = 64; g.Kb = 64;
    g.scale = 0.125f; g.Z2 = 12; g.Z3 = KH;
    gemm_k<4, true><<<ggrid(325, 336, BB * 12 * KH), 256, 0, stream>>>(g); }
  softmax_k<<<(BB * 12 * KH * 325 + 3) / 4, 128, 0, stream>>>(scores3, BB * 12 * KH * 325, 336);
  { GP g = gp0(); g.A = scores3; g.lda = 336;
    g.sA1 = 12LL * KH * 325 * 336; g.sA2 = (long long)KH * 325 * 336; g.sA3 = 325 * 336;
    g.B = qkv3 + 64; g.ldb = 1024;
    g.sB1 = 12LL * NNODE * 1024; g.sB2 = (long long)NNODE * 1024; g.sB3 = 128;
    g.C = attC3; g.ldc = 512; g.sC1 = 12LL * NNODE * 512; g.sC2 = (long long)NNODE * 512; g.sC3 = 64;
    g.RES = htb; g.ldr = 512; g.sR1 = g.sC1; g.sR2 = g.sC2; g.sR3 = 64;
    g.M = 325; g.N = 64; g.Nb = 64; g.Kdim = 336; g.Ka = 336; g.Kb = 325;
    g.Z2 = 12; g.Z3 = KH;
    gemm_k<5, false><<<ggrid(325, 64, BB * 12 * KH), 256, 0, stream>>>(g); }
  { GP g = gp0(); g.A = attC3; g.lda = 512; g.B = WH + OH_SFFN1; g.ldb = 512;
    g.bias = (const float*)d_in[38]; g.C = hid; g.ldc = 512;
    g.M = TOK_DEC; g.N = 512; g.Nb = 512; g.Kdim = 512; g.Ka = 512; g.Kb = 512;
    gemm_k<1, false><<<ggrid(TOK_DEC, 512, 1), 256, 0, stream>>>(g); }
  { GP g = gp0(); g.A = hid; g.lda = 512; g.B = WH + OH_SFFN2; g.ldb = 512;
    g.bias = (const float*)d_in[40]; g.C = hs; g.ldc = 512;
    g.RES = attC3; g.ldr = 512;
    g.M = TOK_DEC; g.N = 512; g.Nb = 512; g.Kdim = 512; g.Ka = 512; g.Kb = 512;
    gemm_k<6, false><<<ggrid(TOK_DEC, 512, 1), 256, 0, stream>>>(g); }
  add_k<<<(TOK_DEC * DD + 255) / 256, 256, 0, stream>>>(hbuf, hs, hfin, TOK_DEC * DD);

  // ================= OUTPUT HEAD =================
  { GP g = gp0(); g.A = hfin; g.lda = 512; g.B = WH + OH_OUT1; g.ldb = 512;
    g.bias = (const float*)d_in[42]; g.C = o1; g.ldc = 512;
    g.M = TOK_DEC; g.N = 512; g.Nb = 512; g.Kdim = 512; g.Ka = 512; g.Kb = 512;
    gemm_k<1, false><<<ggrid(TOK_DEC, 512, 1), 256, 0, stream>>>(g); }
  out2_k<<<(TOK_DEC + 7) / 8, 256, 0, stream>>>(o1, (const float*)d_in[43],
                                                (const float*)d_in[44],
                                                (float*)d_out, TOK_DEC);
}
